// CRF_17746804867875
// MI455X (gfx1250) — matrix operand order repaired and checked by reading, not compiled, not run on hardware
//
#include <hip/hip_runtime.h>

// Problem constants (from reference): B=512, L=1024, T=48
static constexpr int NB = 512;
static constexpr int SL = 1024;
static constexpr int NT = 48;
static constexpr int KP = 64;   // K padded to 64 for 2x wmma_16x16x32

typedef __attribute__((ext_vector_type(16))) _Float16 v16h;
typedef __attribute__((ext_vector_type(8)))  _Float16 v8h;
typedef __attribute__((ext_vector_type(8)))  float    v8f;

// ---------------------------------------------------------------------------
// Forward algorithm: one wave handles 16 batch rows.
// alpha carried as (Mlog, w) with alpha[j] = Mlog + log(w[j]), max(w)=1.
// Per step:  u = w x E  (E = exp(transitions), f16 WMMA, f32 acc),
//            t[j] = u[j]*exp(feat[t][j]);  m2=max_j; w=t/m2; Mlog+=log(m2)
// Elementwise phase is split across the lane pair (l, l+16): each lane owns
// 24 of the 48 columns of one batch row; row max combined via shfl_xor(16).
// feats loads are software-pipelined one step ahead (fv regs), prefetch two.
// ---------------------------------------------------------------------------
__global__ __launch_bounds__(32) void crf_forward_kernel(
    const float* __restrict__ feats,   // [NB][SL][NT]
    const float* __restrict__ trans,   // [NT][NT]
    const float* __restrict__ startt,  // [NT]
    const float* __restrict__ endt,    // [NT]
    const int*   __restrict__ mask,    // [NB][SL]
    float*       __restrict__ fwd_out) // [NB]
{
  __shared__ __align__(16) _Float16 sE[NT][KP];  // sE[n][k] = exp(trans[k][n]); k>=NT -> 0
  __shared__ __align__(16) _Float16 sW[16][KP];  // A-matrix rows (w), K-padded with zeros
  __shared__ __align__(16) float    sU[16][NT];  // f32 matmul result

  const int lane   = threadIdx.x;
  const int rowblk = blockIdx.x * 16;
  const int r      = lane & 15;          // batch row within tile / WMMA M
  const int half   = lane >> 4;          // 0: cols 0-23, 1: cols 24-47 (row phase)
  const int cbase  = half * 24;
  const int b      = rowblk + r;

  // ---- init B-matrix source: transposed exp(transitions), zero-padded K ----
  for (int idx = lane; idx < NT * KP; idx += 32) {
    int n = idx / KP, k = idx % KP;
    sE[n][k] = (k < NT) ? (_Float16)__expf(trans[k * NT + n]) : (_Float16)0.f;
  }

  // sequence length for this lane's row (mask is monotone 1..10..0)
  int len = 0;
  {
    const int* mrow = mask + (size_t)b * SL;
    for (int tt = 0; tt < SL; ++tt) len += mrow[tt];
  }

  // per-lane pointer to its 24-column segment of feats row b at time t
  const float* fbase = feats + (size_t)b * SL * NT + cbase;

  // ---- alpha0 = start + feats[:,0,:]  ->  (Mlog, w) -----------------------
  float Mlog;
  {
    const float4* p = (const float4*)fbase;   // t = 0
    float a[24]; float m = -1e30f;
    #pragma unroll
    for (int i = 0; i < 6; ++i) {
      float4 q = p[i];
      a[4*i+0] = startt[cbase + 4*i + 0] + q.x;
      a[4*i+1] = startt[cbase + 4*i + 1] + q.y;
      a[4*i+2] = startt[cbase + 4*i + 2] + q.z;
      a[4*i+3] = startt[cbase + 4*i + 3] + q.w;
    }
    #pragma unroll
    for (int j = 0; j < 24; ++j) m = fmaxf(m, a[j]);
    m = fmaxf(m, __shfl_xor(m, 16, 32));    // row max across the lane pair
    Mlog = m;
    #pragma unroll
    for (int c = 0; c < 3; ++c) {
      v8h h;
      #pragma unroll
      for (int i = 0; i < 8; ++i) h[i] = (_Float16)__expf(a[8*c + i] - m);
      *(v8h*)&sW[r][cbase + 8*c] = h;
    }
    if (half == 1) {                        // zero-pad K = 48..63
      v8h z = {};
      *(v8h*)&sW[r][48] = z;
      *(v8h*)&sW[r][56] = z;
    }
  }
  __syncthreads();

  // ---- loop-invariant B fragments: 2 K-fragments x 3 N-tiles --------------
  // 16-bit B 32x16 layout: N = lane&15; lanes 0-15 hold K={0..7,16..23}, lanes 16-31
  // hold K={8..15,24..31} (2 halves per VGPR), matching the A fragment
  const int ncol  = lane & 15;
  const int khalf = (lane < 16) ? 0 : 8;   // B-fragment K base within 16
  const int koff  = (lane < 16) ? 0 : 8;   // A-fragment K base within 16
  v16h Bf[2][3];
  #pragma unroll
  for (int kf = 0; kf < 2; ++kf) {
    #pragma unroll
    for (int nt = 0; nt < 3; ++nt) {
      const _Float16* p = &sE[nt * 16 + ncol][kf * 32 + khalf];
      v8h lo = *(const v8h*)p;
      v8h hi = *(const v8h*)(p + 16);
      #pragma unroll
      for (int i = 0; i < 8; ++i) { Bf[kf][nt][i] = lo[i]; Bf[kf][nt][8 + i] = hi[i]; }
    }
  }

  // ---- software-pipelined feats segment for step t=1 ----------------------
  float4 fv[6];
  {
    const float4* p = (const float4*)(fbase + (size_t)1 * NT);
    #pragma unroll
    for (int i = 0; i < 6; ++i) fv[i] = p[i];
  }

  for (int t = 1; t < SL; ++t) {
    // ---- matmul phase (whole wave, EXEC all-1): U = W x E -----------------
    // 16-bit A 16x32 layout: lane<16: M=lane, K={koff..+7, 16+koff..+7}
    v16h A0, A1;
    {
      const _Float16* p0 = &sW[r][koff];
      const _Float16* p1 = &sW[r][32 + koff];
      v8h a0lo = *(const v8h*)p0;
      v8h a0hi = *(const v8h*)(p0 + 16);
      v8h a1lo = *(const v8h*)p1;
      v8h a1hi = *(const v8h*)(p1 + 16);
      #pragma unroll
      for (int i = 0; i < 8; ++i) {
        A0[i] = a0lo[i]; A0[8 + i] = a0hi[i];
        A1[i] = a1lo[i]; A1[8 + i] = a1hi[i];
      }
    }
    v8f acc0 = {}, acc1 = {}, acc2 = {};
    acc0 = __builtin_amdgcn_wmma_f32_16x16x32_f16(false, A0, false, Bf[0][0], (short)0, acc0, false, false);
    acc1 = __builtin_amdgcn_wmma_f32_16x16x32_f16(false, A0, false, Bf[0][1], (short)0, acc1, false, false);
    acc2 = __builtin_amdgcn_wmma_f32_16x16x32_f16(false, A0, false, Bf[0][2], (short)0, acc2, false, false);
    acc0 = __builtin_amdgcn_wmma_f32_16x16x32_f16(false, A1, false, Bf[1][0], (short)0, acc0, false, false);
    acc1 = __builtin_amdgcn_wmma_f32_16x16x32_f16(false, A1, false, Bf[1][1], (short)0, acc1, false, false);
    acc2 = __builtin_amdgcn_wmma_f32_16x16x32_f16(false, A1, false, Bf[1][2], (short)0, acc2, false, false);

    // D layout: lanes 0-15 VGPR k -> M=k, N=lane; lanes 16-31 -> M=k+8
    const int rb = (lane < 16) ? 0 : 8;
    #pragma unroll
    for (int k = 0; k < 8; ++k) {
      sU[rb + k][ncol]      = acc0[k];
      sU[rb + k][16 + ncol] = acc1[k];
      sU[rb + k][32 + ncol] = acc2[k];
    }
    // warm L2 for step t+2 while the D spill drains
    if (t + 2 < SL) __builtin_prefetch(fbase + (size_t)(t + 2) * NT, 0, 3);
    __syncthreads();

    // ---- row phase: lane pair (r, r+16) splits the 48 columns -------------
    if (t < len) {                         // masked rows stay frozen
      float tv[24]; float m2 = 0.f;
      #pragma unroll
      for (int i = 0; i < 6; ++i) {
        tv[4*i+0] = sU[r][cbase + 4*i + 0] * __expf(fv[i].x);
        tv[4*i+1] = sU[r][cbase + 4*i + 1] * __expf(fv[i].y);
        tv[4*i+2] = sU[r][cbase + 4*i + 2] * __expf(fv[i].z);
        tv[4*i+3] = sU[r][cbase + 4*i + 3] * __expf(fv[i].w);
      }
      #pragma unroll
      for (int j = 0; j < 24; ++j) m2 = fmaxf(m2, tv[j]);
      m2 = fmaxf(m2, __shfl_xor(m2, 16, 32));   // pair-uniform predicate
      const float inv = 1.0f / m2;
      #pragma unroll
      for (int c = 0; c < 3; ++c) {
        v8h h;
        #pragma unroll
        for (int i = 0; i < 8; ++i) h[i] = (_Float16)(tv[8*c + i] * inv);
        *(v8h*)&sW[r][cbase + 8*c] = h;
      }
      Mlog += __logf(m2);
    }

    // issue next step's feats loads; latency overlaps next matmul phase
    if (t + 1 < SL) {
      const float4* p = (const float4*)(fbase + (size_t)(t + 1) * NT);
      #pragma unroll
      for (int i = 0; i < 6; ++i) fv[i] = p[i];
    }
    __syncthreads();
  }

  // fwd[b] = Mlog + log(sum_j w[j]*exp(end[j]))   (lanes 0-15 only)
  if (lane < 16) {
    float s = 0.f;
    #pragma unroll
    for (int j = 0; j < NT; ++j) s += (float)sW[r][j] * __expf(endt[j]);
    fwd_out[b] = Mlog + __logf(s);
  }
}

// ---------------------------------------------------------------------------
// Gold path score: one block per batch row, deterministic tree reduction.
// ---------------------------------------------------------------------------
__global__ __launch_bounds__(256) void crf_gold_kernel(
    const float* __restrict__ feats, const float* __restrict__ trans,
    const float* __restrict__ startt, const float* __restrict__ endt,
    const int* __restrict__ tags, const int* __restrict__ mask,
    float* __restrict__ gold_out)
{
  __shared__ float redf[256];
  __shared__ int   redi[256];
  const int b   = blockIdx.x;
  const int tid = threadIdx.x;
  const int*   tg = tags + (size_t)b * SL;
  const int*   mk = mask + (size_t)b * SL;
  const float* fb = feats + (size_t)b * SL * NT;

  float acc = 0.f;
  int   lsum = 0;
  for (int t = tid; t < SL; t += 256) {
    lsum += mk[t];
    if (t < SL - 1) {
      acc += trans[tg[t] * NT + tg[t + 1]] * (float)mk[t + 1]
           + fb[(size_t)t * NT + tg[t]]   * (float)mk[t];
    }
  }
  redf[tid] = acc; redi[tid] = lsum;
  __syncthreads();
  for (int off = 128; off > 0; off >>= 1) {
    if (tid < off) { redf[tid] += redf[tid + off]; redi[tid] += redi[tid + off]; }
    __syncthreads();
  }
  if (tid == 0) {
    const int len = redi[0];
    const int last_tag = tg[len - 1];
    float g = redf[0] + startt[tg[0]] + endt[last_tag]
            + fb[(size_t)(SL - 1) * NT + last_tag] * (float)mk[SL - 1];
    gold_out[b] = g;
  }
}

// ---------------------------------------------------------------------------
// Final deterministic reduction: out = sum_b (fwd[b] - gold[b])
// ---------------------------------------------------------------------------
__global__ __launch_bounds__(256) void crf_final_reduce(
    const float* __restrict__ fwd, const float* __restrict__ gold,
    float* __restrict__ out)
{
  __shared__ float red[256];
  const int tid = threadIdx.x;
  float s = (fwd[tid] - gold[tid]) + (fwd[tid + 256] - gold[tid + 256]);
  red[tid] = s;
  __syncthreads();
  for (int off = 128; off > 0; off >>= 1) {
    if (tid < off) red[tid] += red[tid + off];
    __syncthreads();
  }
  if (tid == 0) out[0] = red[0];
}

extern "C" void kernel_launch(void* const* d_in, const int* in_sizes, int n_in,
                              void* d_out, int out_size, void* d_ws, size_t ws_size,
                              hipStream_t stream) {
  const float* feats  = (const float*)d_in[0];
  const float* trans  = (const float*)d_in[1];
  const float* startt = (const float*)d_in[2];
  const float* endt   = (const float*)d_in[3];
  const int*   tags   = (const int*)d_in[4];
  const int*   mask   = (const int*)d_in[5];

  float* fwd  = (float*)d_ws;       // [NB]
  float* gold = fwd + NB;           // [NB]
  float* out  = (float*)d_out;      // [1]

  crf_forward_kernel<<<NB / 16, 32, 0, stream>>>(feats, trans, startt, endt, mask, fwd);
  crf_gold_kernel<<<NB, 256, 0, stream>>>(feats, trans, startt, endt, tags, mask, gold);
  crf_final_reduce<<<1, 256, 0, stream>>>(fwd, gold, out);
}
